// MultiheadAttention_69045894250510
// MI455X (gfx1250) — matrix operand order repaired and checked by reading, not compiled, not run on hardware
//
#include <hip/hip_runtime.h>

typedef __bf16 bf16;
typedef __attribute__((ext_vector_type(16))) __bf16 v16bf;
typedef __attribute__((ext_vector_type(8)))  __bf16 v8bf;
typedef __attribute__((ext_vector_type(8)))  float   v8f;
typedef __attribute__((ext_vector_type(4)))  float   f32x4;

#define DEVINL __device__ __forceinline__

constexpr int T_SEQ = 1000;
constexpr int BATCH = 8;
constexpr int EMB   = 1024;
constexpr int NH    = 16;
constexpr int HD    = 64;
constexpr int TP    = 1024;             // padded sequence length
constexpr int M_TOT = T_SEQ * BATCH;    // 8000 rows for the projection GEMMs

// ---------------- WMMA fragment helpers (CDNA5 16x16x32 bf16) ----------------
// A (16x32, MxK): lane m = lane&15; VGPR0-3 hold K = ((lane>>4)*8)+0..7,
//                 VGPR4-7 hold those +16  -> two b128 loads per lane.
DEVINL v16bf frag_a16(const bf16* p0, int stride, int lane) {
  const bf16* p = p0 + (lane & 15) * stride + ((lane >> 4) << 3);
  v8bf lo = *(const v8bf*)(p);
  v8bf hi = *(const v8bf*)(p + 16);
  return __builtin_shufflevector(lo, hi, 0,1,2,3,4,5,6,7,8,9,10,11,12,13,14,15);
}
// B (32x16, KxN) from a transposed LDS tile Bt[n][k] (k contiguous):
// lane n = lane&15; K = ((lane>>4)*8)+0..7 and those +16 (same as A).
DEVINL v16bf frag_b16(const bf16* p0, int stride, int lane) {
  const bf16* p = p0 + (lane & 15) * stride + ((lane >> 4) << 3);
  v8bf lo = *(const v8bf*)(p);
  v8bf hi = *(const v8bf*)(p + 16);
  return __builtin_shufflevector(lo, hi, 0,1,2,3,4,5,6,7,8,9,10,11,12,13,14,15);
}
DEVINL v8f wmma_bf16(v16bf a, v16bf b, v8f c) {
  return __builtin_amdgcn_wmma_f32_16x16x32_bf16(
      /*neg_a=*/false, a, /*neg_b=*/false, b,
      /*c_mod=*/(short)0, c, /*reuse_a=*/false, /*reuse_b=*/false);
}

// ---------------- Projection GEMM: out(8000x1024) = x @ W^T + b --------------
// MODE 0: Q  (f32 in, *0.125, RoPE, -> qbf  (BH, TP, HD) bf16)
// MODE 1: K  (f32 in, RoPE,          -> kbf  (BH, TP, HD) bf16)
// MODE 2: V  (f32 in,                -> vtb  (BH, HD, TP) bf16, transposed)
// MODE 3: O  (bf16 attn in,          -> d_out (T,B,E) f32)
template<int MODE>
__global__ __launch_bounds__(256) void proj_kernel(
    const void* __restrict__ xin, const float* __restrict__ W,
    const float* __restrict__ bias, void* __restrict__ outp)
{
  constexpr int AST = 40;                         // LDS k-stride (32+8), 80B rows
  __shared__ __align__(16) unsigned char smem[128 * 68 * 4];
  bf16*  As = (bf16*)smem;                        // 128 x 40 bf16
  bf16*  Bs = (bf16*)(smem + 128 * AST * 2);      // 64  x 40 bf16 (rows of W)
  float* Cs = (float*)smem;                       // reused: 128 x 68 f32

  const int tid  = threadIdx.x;
  const int lane = tid & 31;
  const int w    = tid >> 5;
  const int wm   = (w >> 1) * 32;                 // wave row offset in tile
  const int wn   = (w & 1) * 32;                  // wave col offset in tile
  const int i0   = blockIdx.x * 128;
  const int j0   = blockIdx.y * 64;

  v8f acc[2][2] = {};

  for (int k0 = 0; k0 < EMB; k0 += 32) {
    __syncthreads();
    { // stage A tile (128 x 32) -> bf16
      const int r  = tid >> 1;
      const int kq = (tid & 1) * 16;
      const int gi = i0 + r;
      bf16* dst = As + r * AST + kq;
      if constexpr (MODE < 3) {
        if (gi < M_TOT) {
          const float* s = (const float*)xin + (size_t)gi * EMB + k0 + kq;
          f32x4 f0 = *(const f32x4*)(s + 0);
          f32x4 f1 = *(const f32x4*)(s + 4);
          f32x4 f2 = *(const f32x4*)(s + 8);
          f32x4 f3 = *(const f32x4*)(s + 12);
          #pragma unroll
          for (int e = 0; e < 4; ++e) {
            dst[e] = (bf16)f0[e]; dst[4+e] = (bf16)f1[e];
            dst[8+e] = (bf16)f2[e]; dst[12+e] = (bf16)f3[e];
          }
        } else {
          #pragma unroll
          for (int e = 0; e < 16; ++e) dst[e] = (bf16)0.f;
        }
      } else {
        if (gi < M_TOT) {
          const bf16* s = (const bf16*)xin + (size_t)gi * EMB + k0 + kq;
          *(v8bf*)dst       = *(const v8bf*)s;
          *(v8bf*)(dst + 8) = *(const v8bf*)(s + 8);
        } else {
          #pragma unroll
          for (int e = 0; e < 16; ++e) dst[e] = (bf16)0.f;
        }
      }
    }
    { // stage B tile: rows j0..j0+63 of W, k0..k0+31 (Bt[n][k] == W[n][k])
      const int r  = tid >> 2;
      const int kq = (tid & 3) * 8;
      const float* s = W + (size_t)(j0 + r) * EMB + k0 + kq;
      f32x4 f0 = *(const f32x4*)(s);
      f32x4 f1 = *(const f32x4*)(s + 4);
      bf16* dst = Bs + r * AST + kq;
      #pragma unroll
      for (int e = 0; e < 4; ++e) { dst[e] = (bf16)f0[e]; dst[4+e] = (bf16)f1[e]; }
    }
    __syncthreads();

    v16bf a0 = frag_a16(As + (size_t)wm * AST,        AST, lane);
    v16bf a1 = frag_a16(As + (size_t)(wm + 16) * AST, AST, lane);
    v16bf b0 = frag_b16(Bs + (size_t)wn * AST,        AST, lane);
    v16bf b1 = frag_b16(Bs + (size_t)(wn + 16) * AST, AST, lane);
    acc[0][0] = wmma_bf16(a0, b0, acc[0][0]);
    acc[0][1] = wmma_bf16(a0, b1, acc[0][1]);
    acc[1][0] = wmma_bf16(a1, b0, acc[1][0]);
    acc[1][1] = wmma_bf16(a1, b1, acc[1][1]);
  }
  __syncthreads();

  // C layout: lane n = lane&15, VGPR r holds row r + 8*(lane>>4)
  #pragma unroll
  for (int mi = 0; mi < 2; ++mi)
    #pragma unroll
    for (int ni = 0; ni < 2; ++ni)
      #pragma unroll
      for (int r = 0; r < 8; ++r) {
        int m = wm + mi * 16 + ((lane >> 4) << 3) + r;
        int n = wn + ni * 16 + (lane & 15);
        Cs[m * 68 + n] = acc[mi][ni][r];
      }
  __syncthreads();

  // Epilogue: bias (+ scale + RoPE for Q/K). Block's 64 cols == exactly one head.
  #pragma unroll 4
  for (int e = 0; e < 32; ++e) {
    int flat = tid * 32 + e;                // 0..8191 over 128x64 tile
    int m = flat >> 6, n = flat & 63;
    int gi = i0 + m;
    if (gi >= M_TOT) continue;
    int j = j0 + n;
    float v = Cs[m * 68 + n] + bias[j];
    if constexpr (MODE == 0) v *= 0.125f;   // HD^-0.5
    if constexpr (MODE <= 1) {
      float prt = Cs[m * 68 + (n ^ 32)] + bias[j0 + (n ^ 32)];
      if constexpr (MODE == 0) prt *= 0.125f;
      int t = gi >> 3, bb = gi & 7;         // gi = t*B + b
      float inv = __powf(10000.f, -(float)(n & 31) * (1.f / 32.f));
      float sv, cv;
      __sincosf((float)t * inv, &sv, &cv);
      float rot = (n < 32) ? -prt : prt;
      float res = cv * v + sv * rot;
      int bh = bb * NH + (j0 >> 6);
      ((bf16*)outp)[((size_t)bh * TP + t) * HD + n] = (bf16)res;
    } else if constexpr (MODE == 2) {
      int t = gi >> 3, bb = gi & 7;
      int bh = bb * NH + (j0 >> 6);
      ((bf16*)outp)[((size_t)bh * HD + n) * TP + t] = (bf16)v;   // V^T
    } else {
      ((float*)outp)[(size_t)gi * EMB + j] = v;
    }
  }
}

// ---------------- Flash attention: per (b,h), online softmax --------------
// grid = (8 q-blocks of 128 rows, 128 head-batches); 8 waves, 16 q-rows/wave.
__global__ __launch_bounds__(256) void flash_attn_kernel(
    const bf16* __restrict__ qbf, const bf16* __restrict__ kbf,
    const bf16* __restrict__ vtb, bf16* __restrict__ attn)
{
  constexpr int KST = 72;                    // 64+8 bf16 => 144B rows (16B mult)
  __shared__ __align__(16) bf16 Ks[64 * KST];        // K tile rows tk x d
  __shared__ __align__(16) bf16 Vs[64 * KST];        // V^T tile rows d x tk
  __shared__ __align__(16) bf16 Ps[8 * 16 * KST];    // per-wave P transpose buf

  const int tid   = threadIdx.x;
  const int lane  = tid & 31;
  const int w     = tid >> 5;
  const int bh    = blockIdx.y;
  const int q0    = blockIdx.x * 128 + w * 16;
  const int ncol  = lane & 15;
  const int rowhi = (lane >> 4) << 3;

  const bf16* qbase = qbf + ((size_t)bh * TP + q0) * HD;
  v16bf qa0 = frag_a16(qbase,      HD, lane);        // k = 0..31
  v16bf qa1 = frag_a16(qbase + 32, HD, lane);        // k = 32..63

  v8f o[4] = {};
  float mrow[8], lrow[8];
  #pragma unroll
  for (int r = 0; r < 8; ++r) { mrow[r] = -1e30f; lrow[r] = 0.f; }

  bf16* pw = Ps + (size_t)w * 16 * KST;

  for (int kb = 0; kb < TP / 64; ++kb) {
    const int kt0 = kb * 64;
    __syncthreads();
    { // stage K (64 x 64) and V^T (64 x 64), 16 bf16 per thread each
      const int r  = tid >> 2;
      const int kq = (tid & 3) * 16;
      const bf16* sk = kbf + ((size_t)bh * TP + kt0 + r) * HD + kq;
      *(v8bf*)(Ks + r * KST + kq)     = *(const v8bf*)sk;
      *(v8bf*)(Ks + r * KST + kq + 8) = *(const v8bf*)(sk + 8);
      const bf16* sv = vtb + ((size_t)bh * HD + r) * TP + kt0 + kq;
      *(v8bf*)(Vs + r * KST + kq)     = *(const v8bf*)sv;
      *(v8bf*)(Vs + r * KST + kq + 8) = *(const v8bf*)(sv + 8);
    }
    __syncthreads();

    // S = Q @ K^T : 4 col-tiles x 2 k-steps
    v8f sacc[4] = {};
    #pragma unroll
    for (int kk = 0; kk < 64; kk += 32) {
      v16bf a = (kk == 0) ? qa0 : qa1;
      #pragma unroll
      for (int j = 0; j < 4; ++j) {
        v16bf b = frag_b16(Ks + (size_t)j * 16 * KST + kk, KST, lane);
        sacc[j] = wmma_bf16(a, b, sacc[j]);
      }
    }
    // mask padded key columns
    #pragma unroll
    for (int j = 0; j < 4; ++j)
      if (kt0 + j * 16 + ncol >= T_SEQ) {
        #pragma unroll
        for (int r = 0; r < 8; ++r) sacc[j][r] = -1e30f;
      }

    // online softmax: rows live in one VGPR across a 16-lane half
    float alpha[8];
    #pragma unroll
    for (int r = 0; r < 8; ++r) {
      float mx = fmaxf(fmaxf(sacc[0][r], sacc[1][r]), fmaxf(sacc[2][r], sacc[3][r]));
      #pragma unroll
      for (int off = 1; off < 16; off <<= 1) mx = fmaxf(mx, __shfl_xor(mx, off, 32));
      float mn = fmaxf(mrow[r], mx);
      alpha[r] = __expf(mrow[r] - mn);
      mrow[r] = mn;
      float sum = 0.f;
      #pragma unroll
      for (int j = 0; j < 4; ++j) {
        float p = __expf(sacc[j][r] - mn);
        sacc[j][r] = p;
        sum += p;
      }
      #pragma unroll
      for (int off = 1; off < 16; off <<= 1) sum += __shfl_xor(sum, off, 32);
      lrow[r] = lrow[r] * alpha[r] + sum;
    }

    // P: C-layout -> LDS -> A-layout
    #pragma unroll
    for (int j = 0; j < 4; ++j)
      #pragma unroll
      for (int r = 0; r < 8; ++r)
        pw[(rowhi + r) * KST + j * 16 + ncol] = (bf16)sacc[j][r];
    __syncthreads();

    // O = alpha*O + P @ V
    #pragma unroll
    for (int j = 0; j < 4; ++j)
      #pragma unroll
      for (int r = 0; r < 8; ++r) o[j][r] *= alpha[r];
    #pragma unroll
    for (int kk = 0; kk < 64; kk += 32) {
      v16bf pa = frag_a16(pw + kk, KST, lane);
      #pragma unroll
      for (int j = 0; j < 4; ++j) {
        v16bf b = frag_b16(Vs + (size_t)j * 16 * KST + kk, KST, lane);
        o[j] = wmma_bf16(pa, b, o[j]);
      }
    }
  }

  // normalize and write attn in (T, B, E) layout (bf16 for final GEMM)
  const int b = bh >> 4;
  const int h = bh & 15;
  #pragma unroll
  for (int r = 0; r < 8; ++r) {
    const int t = q0 + rowhi + r;
    if (t >= T_SEQ) continue;
    const float invl = 1.f / lrow[r];
    #pragma unroll
    for (int j = 0; j < 4; ++j)
      attn[((size_t)t * BATCH + b) * EMB + h * HD + j * 16 + ncol] =
          (bf16)(o[j][r] * invl);
  }
}

// ---------------- host launcher ----------------
extern "C" void kernel_launch(void* const* d_in, const int* in_sizes, int n_in,
                              void* d_out, int out_size, void* d_ws, size_t ws_size,
                              hipStream_t stream) {
  (void)in_sizes; (void)n_in; (void)out_size; (void)ws_size;
  const float* query = (const float*)d_in[0];
  const float* key_  = (const float*)d_in[1];
  const float* value = (const float*)d_in[2];
  const float* q_w   = (const float*)d_in[3];
  const float* k_w   = (const float*)d_in[4];
  const float* v_w   = (const float*)d_in[5];
  const float* o_w   = (const float*)d_in[6];
  const float* q_b   = (const float*)d_in[7];
  const float* k_b   = (const float*)d_in[8];
  const float* v_b   = (const float*)d_in[9];
  const float* o_b   = (const float*)d_in[10];

  const size_t headElems = (size_t)BATCH * NH * TP * HD;  // 8,388,608 bf16
  bf16* qbf  = (bf16*)d_ws;
  bf16* kbf  = qbf + headElems;
  bf16* vtb  = kbf + headElems;
  bf16* attn = vtb + headElems;   // (T*B rows, E) bf16; total ws use = 64 MB

  dim3 blk(256);
  dim3 gp(63, 16);                // ceil(8000/128) x (1024/64)
  proj_kernel<0><<<gp, blk, 0, stream>>>(query, q_w, q_b, qbf);
  proj_kernel<1><<<gp, blk, 0, stream>>>(key_,  k_w, k_b, kbf);
  proj_kernel<2><<<gp, blk, 0, stream>>>(value, v_w, v_b, vtb);
  flash_attn_kernel<<<dim3(8, 128), blk, 0, stream>>>(qbf, kbf, vtb, attn);
  proj_kernel<3><<<gp, blk, 0, stream>>>(attn, o_w, o_b, d_out);
}
